// Collection_Unit_wAttention_14259291423186
// MI455X (gfx1250) — matrix operand order repaired and checked by reading, not compiled, not run on hardware
//
#include <hip/hip_runtime.h>
#include <stdint.h>

// ---------------------------------------------------------------------------
// GAT 3-layer pipeline for MI455X (gfx1250, wave32, WMMA, async LDS-staged B)
// ---------------------------------------------------------------------------

#define NEG_SLOPE 0.2f

#ifndef __has_builtin
#define __has_builtin(x) 0
#endif

typedef __attribute__((ext_vector_type(16))) __bf16        v16bf;
typedef __attribute__((ext_vector_type(8)))  float         v8f;

union FragBF {
    v16bf v;
    uint32_t u[8];
};

// ---- f32 -> bf16 (round to nearest even, keep inf/nan) --------------------
__device__ __forceinline__ uint16_t f32_to_bf16(float f) {
    uint32_t x = __float_as_uint(f);
    uint32_t r = x + 0x7FFFu + ((x >> 16) & 1u);
    if ((x & 0x7F800000u) == 0x7F800000u) r = x;   // inf / nan passthrough
    return (uint16_t)(r >> 16);
}

// ---- monotone float<->uint key for atomicMax on floats --------------------
__device__ __forceinline__ unsigned int fkey(float f) {
    unsigned int b = __float_as_uint(f);
    return (b & 0x80000000u) ? ~b : (b | 0x80000000u);
}
__device__ __forceinline__ float fkey_decode(unsigned int k) {
    unsigned int b = (k & 0x80000000u) ? (k ^ 0x80000000u) : ~k;
    return __uint_as_float(b);
}
#define FKEY_NEG_INF 0x007FFFFFu   // fkey(-inf)

// ---- async global->LDS (CDNA5) with compile-safe fallback -----------------
// Probed signature (from round-2 diagnostic): param0 is
//   'int __attribute__((vector_size(16))) __device__ *'  (AS1, non-const)
typedef int v4i_t __attribute__((vector_size(16)));

#if __has_builtin(__builtin_amdgcn_global_load_async_to_lds_b128)
#define HAVE_ASYNC_LDS 1
typedef __attribute__((address_space(1))) v4i_t* as1_v4i;
typedef __attribute__((address_space(3))) v4i_t* as3_v4i;
#endif

__device__ __forceinline__ void copy16_g2lds(const uint16_t* __restrict__ src,
                                             uint16_t* __restrict__ dst_lds) {
#if defined(HAVE_ASYNC_LDS)
    __builtin_amdgcn_global_load_async_to_lds_b128((as1_v4i)src, (as3_v4i)dst_lds,
                                                   0, 0);
#else
    *(uint4*)dst_lds = *(const uint4*)src;
#endif
}

__device__ __forceinline__ void wait_async_copies() {
#if defined(HAVE_ASYNC_LDS)
#if __has_builtin(__builtin_amdgcn_s_wait_asynccnt)
    __builtin_amdgcn_s_wait_asynccnt(0);
#else
    asm volatile("s_wait_asynccnt 0x0" ::: "memory");
#endif
#endif
}

// ===========================================================================
// K0: f32 -> bf16 feature conversion
// ===========================================================================
__global__ void k_convert_bf16(const float* __restrict__ in,
                               uint16_t* __restrict__ out, long long total) {
    long long t = (long long)blockIdx.x * blockDim.x + threadIdx.x;
    long long stride = (long long)gridDim.x * blockDim.x;
    for (; t < total; t += stride) out[t] = f32_to_bf16(in[t]);
}

// ===========================================================================
// K1: transpose W [K=128, Mout] row-major f32 -> Wt [Mout, 128] bf16
// ===========================================================================
__global__ void k_transpose_w(const float* __restrict__ W,
                              uint16_t* __restrict__ Wt, int Mout) {
    int t = blockIdx.x * blockDim.x + threadIdx.x;
    int total = Mout * 128;
    if (t >= total) return;
    int n = t >> 7;
    int k = t & 127;
    Wt[(size_t)n * 128 + k] = f32_to_bf16(W[(size_t)k * Mout + n]);
}

// ===========================================================================
// K2: h = xb @ Wt^T  via v_wmma_f32_16x16x32_bf16, B tile staged in LDS
//   xb : [N,128] bf16 row-major,  Wt : [Mout,128] bf16 (W transposed)
//   h  : [N,Mout] f32
// Block = 256 threads = 8 waves; each wave owns a 16-row stripe.
// ===========================================================================
#define BT_PITCH 136   // 128 + 8 halves padding: kills 16-way LDS bank conflict

__global__ void k_gemm_wmma(const uint16_t* __restrict__ xb,
                            const uint16_t* __restrict__ wt,
                            float* __restrict__ hout,
                            int N, int Mout) {
    __shared__ uint16_t btile[16 * BT_PITCH];   // 16 output cols x K=128 (+pad)

    const int tid  = threadIdx.x;
    const int lane = tid & 31;
    const int wave = tid >> 5;
    const int rowBase = blockIdx.x * 128 + wave * 16;
    const bool waveActive = (rowBase < N);      // wave-uniform

    int r = rowBase + (lane & 15);
    if (r >= N) r = N - 1;                      // clamp A loads (stores guarded)
    if (!waveActive) r = 0;
    const uint16_t* arow = xb + (size_t)r * 128;

    // A fragment (16-bit A 16x32, this code's contiguous k order, same as B):
    //   lanes 0-15: K = kb + 0..15;  lanes 16-31: +16
    const int adelta = (lane < 16) ? 0 : 16;
    // B fragment (16-bit B 32x16): lanes 0-15 K=kb..kb+15, lanes 16-31 +16
    const int khalf  = (lane >> 4) << 4;
    const int ncol   = lane & 15;

    // Hold all 4 K-step A fragments in VGPRs across the whole cb loop.
    FragBF a[4];
#pragma unroll
    for (int s = 0; s < 4; ++s) {
        *(uint4*)&a[s].u[0] = *(const uint4*)(arow + s * 32 + adelta);
        *(uint4*)&a[s].u[4] = *(const uint4*)(arow + s * 32 + adelta + 8);
    }

    // B staging: each of 256 threads copies one 16B segment (16 rows x 16 segs)
    const int crow = tid >> 4;    // 0..15 : output column within tile
    const int cseg = tid & 15;    // 0..15 : 8-half segment within K=128

    for (int cb = 0; cb < Mout; cb += 16) {
        copy16_g2lds(wt + (size_t)(cb + crow) * 128 + cseg * 8,
                     &btile[crow * BT_PITCH + cseg * 8]);
        wait_async_copies();
        __syncthreads();

        if (waveActive) {
            v8f acc = {};
#pragma unroll
            for (int s = 0; s < 4; ++s) {
                const uint16_t* bp = &btile[ncol * BT_PITCH + s * 32 + khalf];
                FragBF b;
                *(uint4*)&b.u[0] = *(const uint4*)bp;
                *(uint4*)&b.u[4] = *(const uint4*)(bp + 8);
                acc = __builtin_amdgcn_wmma_f32_16x16x32_bf16(
                          false, a[s].v, false, b.v, (short)0, acc, false, false);
            }
            // D layout: VGPR v -> M = v (lanes 0-15) / v+8 (lanes 16-31)
            const int rofs = (lane >> 4) << 3;
            if (rowBase + 16 <= N) {             // fast path: full tile, no guard
#pragma unroll
                for (int v = 0; v < 8; ++v)
                    hout[(size_t)(rowBase + v + rofs) * Mout + cb + ncol] = acc[v];
            } else {
#pragma unroll
                for (int v = 0; v < 8; ++v) {
                    int row = rowBase + v + rofs;
                    if (row < N) hout[(size_t)row * Mout + cb + ncol] = acc[v];
                }
            }
        }
        __syncthreads();
    }
}

// ===========================================================================
// K3: attention scores s_src/s_dst [N,H] from h [N,H*cph]
// ===========================================================================
__global__ void k_compute_s(const float* __restrict__ h,
                            const float* __restrict__ a_src,
                            const float* __restrict__ a_dst,
                            float* __restrict__ s_src,
                            float* __restrict__ s_dst,
                            int N, int Hh, int cph) {
    int t = blockIdx.x * blockDim.x + threadIdx.x;
    if (t >= N * Hh) return;
    int n = t / Hh, hd = t % Hh;
    const float* hp = h + (size_t)n * Hh * cph + (size_t)hd * cph;
    const float* as = a_src + hd * cph;
    const float* ad = a_dst + hd * cph;
    float ss = 0.f, sd = 0.f;
    for (int c = 0; c < cph; ++c) { float v = hp[c]; ss += v * as[c]; sd += v * ad[c]; }
    s_src[t] = ss;
    s_dst[t] = sd;
}

// ===========================================================================
// K4: init m (-inf keys) / z (0) ; generic f32 zero fill
// ===========================================================================
__global__ void k_init_mz(unsigned int* __restrict__ mkey,
                          float* __restrict__ z, int total) {
    int t = blockIdx.x * blockDim.x + threadIdx.x;
    if (t >= total) return;
    mkey[t] = FKEY_NEG_INF;
    z[t] = 0.f;
}

__global__ void k_zero_f32(float* __restrict__ p, long long total) {
    long long t = (long long)blockIdx.x * blockDim.x + threadIdx.x;
    long long stride = (long long)gridDim.x * blockDim.x;
    for (; t < total; t += stride) p[t] = 0.f;
}

// ===========================================================================
// K5: segment max of leaky_relu edge scores (monotone-key atomicMax)
// ===========================================================================
__global__ void k_edge_max(const long long* __restrict__ ei,
                           long long Eorig, long long Ecnt,
                           const float4* __restrict__ s_src4,
                           const float4* __restrict__ s_dst4,
                           unsigned int* __restrict__ mkey) {
    long long e = (long long)blockIdx.x * blockDim.x + threadIdx.x;
    if (e >= Ecnt) return;
    long long s, d;
    if (e < Eorig) { s = ei[e]; d = ei[Eorig + e]; } else { s = d = e - Eorig; }
    float4 ss = s_src4[s];
    float4 sd = s_dst4[d];
    float v0 = ss.x + sd.x, v1 = ss.y + sd.y, v2 = ss.z + sd.z, v3 = ss.w + sd.w;
    v0 = (v0 > 0.f) ? v0 : NEG_SLOPE * v0;
    v1 = (v1 > 0.f) ? v1 : NEG_SLOPE * v1;
    v2 = (v2 > 0.f) ? v2 : NEG_SLOPE * v2;
    v3 = (v3 > 0.f) ? v3 : NEG_SLOPE * v3;
    unsigned int* mp = mkey + d * 4;
    atomicMax(mp + 0, fkey(v0));
    atomicMax(mp + 1, fkey(v1));
    atomicMax(mp + 2, fkey(v2));
    atomicMax(mp + 3, fkey(v3));
}

// ===========================================================================
// K6: p = exp(e - m[dst]) ; alpha <- p ; z[dst] += p
// ===========================================================================
__global__ void k_edge_exp(const long long* __restrict__ ei,
                           long long Eorig, long long Ecnt,
                           const float4* __restrict__ s_src4,
                           const float4* __restrict__ s_dst4,
                           const unsigned int* __restrict__ mkey,
                           float4* __restrict__ alpha4,
                           float* __restrict__ z) {
    long long e = (long long)blockIdx.x * blockDim.x + threadIdx.x;
    if (e >= Ecnt) return;
    long long s, d;
    if (e < Eorig) { s = ei[e]; d = ei[Eorig + e]; } else { s = d = e - Eorig; }
    float4 ss = s_src4[s];
    float4 sd = s_dst4[d];
    const unsigned int* mp = mkey + d * 4;
    float v0 = ss.x + sd.x, v1 = ss.y + sd.y, v2 = ss.z + sd.z, v3 = ss.w + sd.w;
    v0 = (v0 > 0.f) ? v0 : NEG_SLOPE * v0;
    v1 = (v1 > 0.f) ? v1 : NEG_SLOPE * v1;
    v2 = (v2 > 0.f) ? v2 : NEG_SLOPE * v2;
    v3 = (v3 > 0.f) ? v3 : NEG_SLOPE * v3;
    float4 p;
    p.x = __expf(v0 - fkey_decode(mp[0]));
    p.y = __expf(v1 - fkey_decode(mp[1]));
    p.z = __expf(v2 - fkey_decode(mp[2]));
    p.w = __expf(v3 - fkey_decode(mp[3]));
    alpha4[e] = p;
    float* zp = z + d * 4;
    atomicAdd(zp + 0, p.x);
    atomicAdd(zp + 1, p.y);
    atomicAdd(zp + 2, p.z);
    atomicAdd(zp + 3, p.w);
}

// ===========================================================================
// K6b: alpha[e,h] = alpha[e,h] / (z[dst,h] + 1e-16) * scale
// (pre-normalizing removes the z gather + division from the hot scatter pass)
// ===========================================================================
__global__ void k_norm_alpha(const long long* __restrict__ ei,
                             long long Eorig, long long Ecnt,
                             const float4* __restrict__ z4,
                             float4* __restrict__ alpha4, float scale) {
    long long e = (long long)blockIdx.x * blockDim.x + threadIdx.x;
    if (e >= Ecnt) return;
    long long d = (e < Eorig) ? ei[Eorig + e] : (e - Eorig);
    float4 p  = alpha4[e];
    float4 zz = z4[d];
    p.x = p.x / (zz.x + 1e-16f) * scale;
    p.y = p.y / (zz.y + 1e-16f) * scale;
    p.z = p.z / (zz.z + 1e-16f) * scale;
    p.w = p.w / (zz.w + 1e-16f) * scale;
    alpha4[e] = p;
}

// ===========================================================================
// K7: weighted scatter-add of messages (alpha already normalized+scaled).
//   out[dst, oc] += h[src, j] * alpha[e, j/cph]
//   concat: oc = j ; mean(L3): oc = j % cph (head-mean folded into scale)
// Thread handles 4 consecutive channels (same head since cph % 4 == 0).
// ===========================================================================
__global__ void k_aggregate(const long long* __restrict__ ei,
                            long long Eorig, long long Ecnt,
                            const float* __restrict__ h,
                            const float* __restrict__ alpha,
                            float* __restrict__ out,
                            int Hh, int HCtot, int cph, int outcols) {
    const int tpe = HCtot >> 2;
    const long long total = Ecnt * (long long)tpe;
    long long stride = (long long)gridDim.x * blockDim.x;
    for (long long t = (long long)blockIdx.x * blockDim.x + threadIdx.x;
         t < total; t += stride) {
        long long e = t / tpe;
        int q = (int)(t - e * tpe);
        long long s, d;
        if (e < Eorig) { s = ei[e]; d = ei[Eorig + e]; } else { s = d = e - Eorig; }
        int j0 = q << 2;
        int hd = j0 / cph;
        float w = alpha[e * Hh + hd];
        float4 hv = *(const float4*)(h + (size_t)s * HCtot + j0);
        int oc = (outcols == HCtot) ? j0 : (j0 % cph);
        float* op = out + (size_t)d * outcols + oc;
        atomicAdd(op + 0, hv.x * w);
        atomicAdd(op + 1, hv.y * w);
        atomicAdd(op + 2, hv.z * w);
        atomicAdd(op + 3, hv.w * w);
    }
}

// ===========================================================================
// K8: epilogue: x_out = (elu?)(agg + b + x_in)
// ===========================================================================
__global__ void k_epilogue(const float* __restrict__ agg,
                           const float* __restrict__ b,
                           const float* __restrict__ xin,
                           float* __restrict__ xout,
                           long long total, int cols, int do_elu) {
    long long t = (long long)blockIdx.x * blockDim.x + threadIdx.x;
    long long stride = (long long)gridDim.x * blockDim.x;
    for (; t < total; t += stride) {
        int j = (int)(t % cols);
        float v = agg[t] + b[j] + xin[t];
        if (do_elu) v = (v > 0.f) ? v : expm1f(v);
        xout[t] = v;
    }
}

// ===========================================================================
// Host-side orchestration
// ===========================================================================
static inline int iceil(long long a, int b) { return (int)((a + b - 1) / b); }

extern "C" void kernel_launch(void* const* d_in, const int* in_sizes, int n_in,
                              void* d_out, int out_size, void* d_ws, size_t ws_size,
                              hipStream_t stream) {
    const int F  = 128;
    const int Hh = 4;
    const long long N     = in_sizes[0] / F;        // 100000
    const long long Eorig = in_sizes[1] / 2;        // 800000
    const long long Ecnt  = Eorig + N;              // + self loops

    const float* x      = (const float*)d_in[0];
    const long long* ei = (const long long*)d_in[1];
    const float* W[3]    = {(const float*)d_in[2], (const float*)d_in[6], (const float*)d_in[10]};
    const float* Asrc[3] = {(const float*)d_in[3], (const float*)d_in[7], (const float*)d_in[11]};
    const float* Adst[3] = {(const float*)d_in[4], (const float*)d_in[8], (const float*)d_in[12]};
    const float* Bias[3] = {(const float*)d_in[5], (const float*)d_in[9], (const float*)d_in[13]};
    const int Mout[3]    = {in_sizes[2] / F, in_sizes[6] / F, in_sizes[10] / F}; // 128,128,512

    // ---- workspace layout -------------------------------------------------
    char* ws = (char*)d_ws;
    size_t off = 0;
    auto alloc = [&](size_t bytes) -> size_t {
        size_t o = off;
        off = (off + bytes + 255) & ~(size_t)255;
        return o;
    };
    uint16_t* xb    = (uint16_t*)(ws + alloc((size_t)N * F * 2));
    uint16_t* Wt    = (uint16_t*)(ws + alloc((size_t)512 * 128 * 2));
    float*    hbuf  = (float*)   (ws + alloc((size_t)N * 512 * 4));
    float*    s_src = (float*)   (ws + alloc((size_t)N * Hh * 4));
    float*    s_dst = (float*)   (ws + alloc((size_t)N * Hh * 4));
    unsigned int* mkey = (unsigned int*)(ws + alloc((size_t)N * Hh * 4));
    float*    zbuf  = (float*)   (ws + alloc((size_t)N * Hh * 4));
    float*    albuf = (float*)   (ws + alloc((size_t)Ecnt * Hh * 4));
    float*    agg   = (float*)   (ws + alloc((size_t)N * F * 4));
    float*    x1    = (float*)   (ws + alloc((size_t)N * F * 4));
    float*    x2    = (float*)   (ws + alloc((size_t)N * F * 4));
    (void)ws_size;

    const float* layer_in[3]  = {x, x1, x2};
    float*       layer_out[3] = {x1, x2, (float*)d_out};

    for (int L = 0; L < 3; ++L) {
        const int Mo  = Mout[L];
        const int cph = Mo / Hh;                     // 32,32,128
        const int concat = (L < 2);
        const long long nodeElems = N * F;

        k_convert_bf16<<<8192, 256, 0, stream>>>(layer_in[L], xb, nodeElems);
        k_transpose_w<<<iceil((long long)Mo * 128, 256), 256, 0, stream>>>(W[L], Wt, Mo);
        k_gemm_wmma<<<iceil(N, 128), 256, 0, stream>>>(xb, Wt, hbuf, (int)N, Mo);
        k_compute_s<<<iceil(N * Hh, 256), 256, 0, stream>>>(hbuf, Asrc[L], Adst[L],
                                                            s_src, s_dst, (int)N, Hh, cph);
        k_init_mz<<<iceil(N * Hh, 256), 256, 0, stream>>>(mkey, zbuf, (int)(N * Hh));
        k_zero_f32<<<8192, 256, 0, stream>>>(agg, nodeElems);
        k_edge_max<<<iceil(Ecnt, 256), 256, 0, stream>>>(ei, Eorig, Ecnt,
                                                         (const float4*)s_src,
                                                         (const float4*)s_dst, mkey);
        k_edge_exp<<<iceil(Ecnt, 256), 256, 0, stream>>>(ei, Eorig, Ecnt,
                                                         (const float4*)s_src,
                                                         (const float4*)s_dst, mkey,
                                                         (float4*)albuf, zbuf);
        float scale = concat ? 1.0f : 1.0f / Hh;
        k_norm_alpha<<<iceil(Ecnt, 256), 256, 0, stream>>>(ei, Eorig, Ecnt,
                                                           (const float4*)zbuf,
                                                           (float4*)albuf, scale);
        k_aggregate<<<65535, 256, 0, stream>>>(ei, Eorig, Ecnt, hbuf, albuf,
                                               agg, Hh, Mo, cph, F);
        k_epilogue<<<8192, 256, 0, stream>>>(agg, Bias[L], layer_in[L], layer_out[L],
                                             nodeElems, F, concat ? 1 : 0);
    }
}
